// AdvancedStateBank_65189013619257
// MI455X (gfx1250) — matrix operand order repaired and checked by reading, not compiled, not run on hardware
//
#include <hip/hip_runtime.h>
#include <hip/hip_bf16.h>
#include <math.h>

typedef __bf16 bf16_t;
typedef __attribute__((ext_vector_type(16))) __bf16 bf16x16;
typedef __attribute__((ext_vector_type(8)))  __bf16 bf16x8;
typedef __attribute__((ext_vector_type(8)))  float  f32x8;

#define D_MODEL 1024
#define BT      2048
#define NLEV    3
#define MAXK_   64

// LDS row pitch in bf16 elements: 32 data + 8 pad = 80B rows (20 banks; keeps
// the 16-lane b128 fragment reads on disjoint bank groups).
#define PITCH   40

// ---------------------------------------------------------------------------
// CDNA5 async global->LDS copy (16B), tracked by ASYNCcnt.
// lds_off = low 32 bits of generic shared pointer (= wave-relative LDS addr).
// ---------------------------------------------------------------------------
__device__ __forceinline__ void async_cp16(unsigned lds_off, const bf16_t* g) {
    asm volatile("global_load_async_to_lds_b128 %0, %1, off"
                 :: "v"(lds_off), "v"(g) : "memory");
}
__device__ __forceinline__ void wait_async0() {
    asm volatile("s_wait_asynccnt 0" ::: "memory");
}
__device__ __forceinline__ unsigned lds_off32(const void* p) {
    return (unsigned)(unsigned long long)p;
}

// ---------------------------------------------------------------------------
// fp32 -> bf16 conversion (vectorized by 4)
// ---------------------------------------------------------------------------
__global__ __launch_bounds__(256)
void cvt_bf16_v4(const float* __restrict__ src, bf16_t* __restrict__ dst, int n4) {
    int i = blockIdx.x * blockDim.x + threadIdx.x;
    if (i < n4) {
        float4 f = ((const float4*)src)[i];
        dst[4 * i + 0] = (bf16_t)f.x;
        dst[4 * i + 1] = (bf16_t)f.y;
        dst[4 * i + 2] = (bf16_t)f.z;
        dst[4 * i + 3] = (bf16_t)f.w;
    }
}

// fp32 [rows,cols] -> bf16 transposed [cols,rows]
__global__ __launch_bounds__(256)
void cvt_bf16_transpose(const float* __restrict__ src, bf16_t* __restrict__ dst,
                        int rows, int cols) {
    int idx = blockIdx.x * blockDim.x + threadIdx.x;
    if (idx < rows * cols) {
        int r = idx / cols, c = idx - r * cols;
        dst[(size_t)c * rows + r] = (bf16_t)src[idx];
    }
}

// ---------------------------------------------------------------------------
// WMMA GEMM with async-LDS double buffering.
//   C[m,n] = epilogue( scale * sum_k A[m,k]*B[n,k] + bias[n] )
// A: [M,K] bf16 row-major, B: [N,K] bf16 row-major.
// Block: 256 thr = 8 waves (2M x 4N). Macro-tile 64(M) x 128(N), K step 32.
// Each wave: 2x2 fragments (32x32), 4 accumulators.
// ---------------------------------------------------------------------------
union ABfrag { bf16x16 v; bf16x8 h[2]; };

__global__ __launch_bounds__(256)
void wmma_gemm_bt(const bf16_t* __restrict__ A, const bf16_t* __restrict__ B,
                  float* __restrict__ C, const float* __restrict__ bias,
                  int N, int K, float scale, int do_gelu) {
    __shared__ __align__(16) bf16_t smA[2][64  * PITCH];   //  2 x 5120B
    __shared__ __align__(16) bf16_t smB[2][128 * PITCH];   //  2 x 10240B

    const int tid  = threadIdx.x;
    const int lane = tid & 31;
    const int wave = tid >> 5;
    const int mw = wave & 1;          // 0..1  -> 32-row slice of 64
    const int nw = wave >> 1;         // 0..3  -> 32-col slice of 128
    const int m0 = blockIdx.x * 64;
    const int n0 = blockIdx.y * 128;
    const int ll = lane & 15;
    const int lh = lane >> 4;

    // staging coords (16B chunks): A tile 64x32 = 256 chunks (1/thread),
    // B tile 128x32 = 512 chunks (2/thread)
    const int arow = tid >> 2, ac = tid & 3;

    const int NK = K / 32;

    // prologue: stage k-step 0 into buffer 0
    {
        async_cp16(lds_off32(&smA[0][arow * PITCH + ac * 8]),
                   A + (size_t)(m0 + arow) * K + ac * 8);
#pragma unroll
        for (int j = 0; j < 2; ++j) {
            int ch = tid + j * 256, br = ch >> 2, bc = ch & 3;
            async_cp16(lds_off32(&smB[0][br * PITCH + bc * 8]),
                       B + (size_t)(n0 + br) * K + bc * 8);
        }
    }
    wait_async0();
    __syncthreads();

    f32x8 acc[2][2] = {};

    for (int kt = 0; kt < NK; ++kt) {
        const int cur = kt & 1;
        if (kt + 1 < NK) {          // prefetch next k-step into other buffer
            const int k0 = (kt + 1) * 32;
            async_cp16(lds_off32(&smA[cur ^ 1][arow * PITCH + ac * 8]),
                       A + (size_t)(m0 + arow) * K + k0 + ac * 8);
#pragma unroll
            for (int j = 0; j < 2; ++j) {
                int ch = tid + j * 256, br = ch >> 2, bc = ch & 3;
                async_cp16(lds_off32(&smB[cur ^ 1][br * PITCH + bc * 8]),
                           B + (size_t)(n0 + br) * K + k0 + bc * 8);
            }
        }

        // fragment reads from LDS (ds_load_b128)
        const bf16_t* As = &smA[cur][0];
        const bf16_t* Bs = &smB[cur][0];
        ABfrag a[2], b[2];
#pragma unroll
        for (int fm = 0; fm < 2; ++fm) {
            const int r = (mw * 32 + fm * 16 + ll) * PITCH;
            a[fm].h[0] = *(const bf16x8*)(As + r + lh * 8);        // K 0-7 / 8-15
            a[fm].h[1] = *(const bf16x8*)(As + r + lh * 8 + 16);   // K 16-23 / 24-31
        }
#pragma unroll
        for (int fn = 0; fn < 2; ++fn) {
            const int r = (nw * 32 + fn * 16 + ll) * PITCH;
            b[fn].h[0] = *(const bf16x8*)(Bs + r + lh * 8);        // K 0-7 / 8-15
            b[fn].h[1] = *(const bf16x8*)(Bs + r + lh * 8 + 16);   // K 16-23 / 24-31
        }

#pragma unroll
        for (int fm = 0; fm < 2; ++fm)
#pragma unroll
            for (int fn = 0; fn < 2; ++fn)
                acc[fm][fn] = __builtin_amdgcn_wmma_f32_16x16x32_bf16(
                    false, a[fm].v, false, b[fn].v, (short)0, acc[fm][fn], false, false);

        if (kt + 1 < NK) wait_async0();
        __syncthreads();
    }

    // epilogue: C layout (f32 16x16): lane half -> row block, col = tile + ll
#pragma unroll
    for (int fm = 0; fm < 2; ++fm) {
#pragma unroll
        for (int fn = 0; fn < 2; ++fn) {
            const int n  = n0 + nw * 32 + fn * 16 + ll;
            const int mb = m0 + mw * 32 + fm * 16 + lh * 8;
            const float bval = bias ? bias[n] : 0.0f;
#pragma unroll
            for (int r = 0; r < 8; ++r) {
                float v = acc[fm][fn][r] * scale + bval;
                if (do_gelu) v = 0.5f * v * (1.0f + erff(v * 0.70710678118654752f));
                C[(size_t)(mb + r) * N + n] = v;
            }
        }
    }
}

// ---------------------------------------------------------------------------
// Router softmax + dynamic-k predictor. One workgroup (256 thr) per query.
// ---------------------------------------------------------------------------
__global__ __launch_bounds__(256)
void router_predictor(const float* __restrict__ h,   // [BT,D] (GELU applied)
                      const float* __restrict__ q,   // [BT,D]
                      const float* __restrict__ rW2, const float* __restrict__ rb2,
                      const float* __restrict__ pW1, const float* __restrict__ pb1,
                      const float* __restrict__ pW2, const float* __restrict__ pb2,
                      const int* __restrict__ maxk_p,
                      float* __restrict__ route,     // [BT,NLEV]
                      int* __restrict__ dynk)        // [BT]
{
    __shared__ float sh[256];
    __shared__ float hid[64];
    __shared__ float logits[NLEV];
    const int qi = blockIdx.x;
    const int t  = threadIdx.x;
    const float* hrow = h + (size_t)qi * D_MODEL;
    const float* qrow = q + (size_t)qi * D_MODEL;

    for (int j = 0; j < NLEV; ++j) {
        float p = 0.f;
        for (int k = t; k < D_MODEL; k += 256) p += hrow[k] * rW2[k * NLEV + j];
        sh[t] = p; __syncthreads();
        for (int s = 128; s > 0; s >>= 1) { if (t < s) sh[t] += sh[t + s]; __syncthreads(); }
        if (t == 0) logits[j] = sh[0] + rb2[j];
        __syncthreads();
    }
    {
        const int col = t >> 2, part = t & 3;
        float p = 0.f;
        for (int k = part; k < D_MODEL; k += 4) p += qrow[k] * pW1[k * 64 + col];
        sh[t] = p; __syncthreads();
        if (part == 0) {
            float v = sh[t] + sh[t + 1] + sh[t + 2] + sh[t + 3] + pb1[col];
            hid[col] = 0.5f * v * (1.0f + erff(v * 0.70710678118654752f));
        }
        __syncthreads();
    }
    if (t == 0) {
        float z = pb2[0];
        for (int c = 0; c < 64; ++c) z += hid[c] * pW2[c];
        float kf = 1.f / (1.f + expf(-z));
        int mk = maxk_p[0];
        int dk = (int)floorf(kf * (float)mk);
        dk = dk < 1 ? 1 : (dk > mk ? mk : dk);
        dynk[qi] = dk;
        float mx = fmaxf(logits[0], fmaxf(logits[1], logits[2]));
        float e0 = expf(logits[0] - mx), e1 = expf(logits[1] - mx), e2 = expf(logits[2] - mx);
        float inv = 1.f / (e0 + e1 + e2);
        route[qi * NLEV + 0] = e0 * inv;
        route[qi * NLEV + 1] = e1 * inv;
        route[qi * NLEV + 2] = e2 * inv;
    }
}

// ---------------------------------------------------------------------------
// Per-query top-k (exact radix select) + softmax + V gather + accumulate.
// ---------------------------------------------------------------------------
template <int PER>
__global__ __launch_bounds__(256)
void topk_gather(const float* __restrict__ scores,  // [BT,S]
                 const float* __restrict__ V,       // [S,D]
                 const int* __restrict__ dynk,
                 const float* __restrict__ route,   // [BT,NLEV]
                 float* __restrict__ out,           // [BT,D] (+=)
                 int S, int level)
{
    __shared__ int   s_cnt[2];
    __shared__ float s_red[256];
    __shared__ int   sel_idx[MAXK_];
    __shared__ float sel_w[MAXK_];
    __shared__ float s_denom;
    const int qi = blockIdx.x;
    const int t  = threadIdx.x;
    const float* srow = scores + (size_t)qi * S;
    const int k_want = dynk[qi];

    float    ls[PER];
    unsigned lk[PER];
    float lmax = -3.402823466e38f;
#pragma unroll
    for (int e = 0; e < PER; ++e) {
        float s = srow[t + e * 256];
        ls[e] = s;
        unsigned u = __float_as_uint(s);
        u ^= (u >> 31) ? 0xFFFFFFFFu : 0x80000000u;
        lk[e] = u;
        lmax = fmaxf(lmax, s);
    }
    s_red[t] = lmax; __syncthreads();
    for (int s = 128; s > 0; s >>= 1) { if (t < s) s_red[t] = fmaxf(s_red[t], s_red[t + s]); __syncthreads(); }
    const float gmax = s_red[0];
    __syncthreads();

    unsigned prefix = 0;
    int kk = k_want;
    for (int bit = 31; bit >= 0; --bit) {
        if (t == 0) s_cnt[0] = 0;
        __syncthreads();
        int c = 0;
#pragma unroll
        for (int e = 0; e < PER; ++e) {
            bool samehi = (((unsigned long long)lk[e]) >> (bit + 1)) ==
                          (((unsigned long long)prefix) >> (bit + 1));
            if (samehi && ((lk[e] >> bit) & 1u)) c++;
        }
        atomicAdd(&s_cnt[0], c);
        __syncthreads();
        int cnt = s_cnt[0];
        if (cnt >= kk) prefix |= (1u << bit); else kk -= cnt;
        __syncthreads();
    }

    if (t == 0) { s_cnt[0] = 0; s_cnt[1] = 0; s_denom = 0.f; }
    __syncthreads();
#pragma unroll
    for (int e = 0; e < PER; ++e) {
        if (lk[e] > prefix) {
            int pos = atomicAdd(&s_cnt[0], 1);
            sel_idx[pos] = t + e * 256;
            sel_w[pos]   = ls[e];
        }
    }
    __syncthreads();
    const int base = s_cnt[0];
#pragma unroll
    for (int e = 0; e < PER; ++e) {
        if (lk[e] == prefix) {
            int pos = atomicAdd(&s_cnt[1], 1);
            if (pos < kk) { sel_idx[base + pos] = t + e * 256; sel_w[base + pos] = ls[e]; }
        }
    }
    __syncthreads();

    const int nsel = k_want;
    if (t < nsel) {
        float w = expf(sel_w[t] - gmax);
        sel_w[t] = w;
        atomicAdd(&s_denom, w);
    }
    __syncthreads();

    const float fac = route[qi * NLEV + level] / s_denom;

    float acc[4] = {0.f, 0.f, 0.f, 0.f};
    for (int j = 0; j < nsel; ++j) {
        const float w = sel_w[j];
        const float* vrow = V + (size_t)sel_idx[j] * D_MODEL;
#pragma unroll
        for (int d = 0; d < 4; ++d) acc[d] += w * vrow[t + d * 256];
    }
    float* orow = out + (size_t)qi * D_MODEL;
#pragma unroll
    for (int d = 0; d < 4; ++d) orow[t + d * 256] += fac * acc[d];
}

// ---------------------------------------------------------------------------
extern "C" void kernel_launch(void* const* d_in, const int* in_sizes, int n_in,
                              void* d_out, int out_size, void* d_ws, size_t ws_size,
                              hipStream_t stream) {
    (void)in_sizes; (void)n_in; (void)out_size; (void)ws_size;

    const float* q    = (const float*)d_in[0];
    const float* Kb[3]  = {(const float*)d_in[1], (const float*)d_in[4], (const float*)d_in[7]};
    const float* Vb[3]  = {(const float*)d_in[2], (const float*)d_in[5], (const float*)d_in[8]};
    const float* sal[3] = {(const float*)d_in[3], (const float*)d_in[6], (const float*)d_in[9]};
    const float* rW1 = (const float*)d_in[10];
    const float* rb1 = (const float*)d_in[11];
    const float* rW2 = (const float*)d_in[12];
    const float* rb2 = (const float*)d_in[13];
    const float* pW1 = (const float*)d_in[14];
    const float* pb1 = (const float*)d_in[15];
    const float* pW2 = (const float*)d_in[16];
    const float* pb2 = (const float*)d_in[17];
    const int*   mk  = (const int*)d_in[18];
    float* out = (float*)d_out;

    // workspace carve (~34 MB)
    char* w = (char*)d_ws;
    bf16_t* q_bf  = (bf16_t*)w;  w += (size_t)BT * D_MODEL * sizeof(bf16_t);
    bf16_t* w_bf  = (bf16_t*)w;  w += (size_t)2048 * D_MODEL * sizeof(bf16_t);  // rW1^T / K-bank
    float*  hbuf  = (float*)w;   w += (size_t)BT * D_MODEL * sizeof(float);
    float*  sc    = (float*)w;   w += (size_t)BT * 2048 * sizeof(float);
    float*  route = (float*)w;   w += (size_t)BT * NLEV * sizeof(float);
    int*    dynk  = (int*)w;

    const float scale = 1.0f / 32.0f;   // 1/sqrt(1024)
    const int S[3] = {2048, 1024, 512};

    {   // q -> bf16
        int n4 = BT * D_MODEL / 4;
        cvt_bf16_v4<<<dim3((n4 + 255) / 256), dim3(256), 0, stream>>>(q, q_bf, n4);
    }
    {   // rW1^T -> bf16
        int n = D_MODEL * D_MODEL;
        cvt_bf16_transpose<<<dim3((n + 255) / 256), dim3(256), 0, stream>>>(rW1, w_bf, D_MODEL, D_MODEL);
    }
    // h = GELU(q @ rW1 + rb1)   [WMMA, async-LDS]
    wmma_gemm_bt<<<dim3(BT / 64, D_MODEL / 128), dim3(256), 0, stream>>>(
        q_bf, w_bf, hbuf, rb1, D_MODEL, D_MODEL, 1.0f, 1);
    // route softmax + dyn_k
    router_predictor<<<dim3(BT), dim3(256), 0, stream>>>(
        hbuf, q, rW2, rb2, pW1, pb1, pW2, pb2, mk, route, dynk);
    // zero output accumulator
    hipMemsetAsync(d_out, 0, (size_t)BT * D_MODEL * sizeof(float), stream);

    for (int i = 0; i < NLEV; ++i) {
        int n4 = S[i] * D_MODEL / 4;
        cvt_bf16_v4<<<dim3((n4 + 255) / 256), dim3(256), 0, stream>>>(Kb[i], w_bf, n4);
        wmma_gemm_bt<<<dim3(BT / 64, S[i] / 128), dim3(256), 0, stream>>>(
            q_bf, w_bf, sc, sal[i], S[i], D_MODEL, scale, 0);
        if (S[i] == 2048)
            topk_gather<8><<<dim3(BT), dim3(256), 0, stream>>>(sc, Vb[i], dynk, route, out, S[i], i);
        else if (S[i] == 1024)
            topk_gather<4><<<dim3(BT), dim3(256), 0, stream>>>(sc, Vb[i], dynk, route, out, S[i], i);
        else
            topk_gather<2><<<dim3(BT), dim3(256), 0, stream>>>(sc, Vb[i], dynk, route, out, S[i], i);
    }
}
